// TorusRouter_49933289783892
// MI455X (gfx1250) — matrix operand order repaired and checked by reading, not compiled, not run on hardware
//
#include <hip/hip_runtime.h>
#include <hip/hip_bf16.h>
#include <math.h>

typedef __attribute__((ext_vector_type(16))) __bf16 v16bf;
typedef __attribute__((ext_vector_type(8)))  float  v8f;

#define N_TOK   16384
#define DMODEL  4096
#define DHALF   2048
#define NEXP    64
#define TOPK    4
#define SCALE_F 2.0f
#define KSTEPS  (DHALF / 32)                 // 64 k-steps of K=32
#define WAVES   4                            // k-split waves per block
#define KS_PER_WAVE (KSTEPS / WAVES)         // 16 k-steps per wave
#define FRAG_ELEMS (KSTEPS * 4 * 32 * 16)    // bf16 elems per packed E matrix = 131072

__device__ inline float fast_exp2(float x) { return __builtin_amdgcn_exp2f(x); }

// ---------------------------------------------------------------------------
// Prep: repack E (fp32 [2048 x 64], row-major) into bf16 B-fragment order:
//   packed[((ks*4 + t)*32 + lane)*16 + j] = E[k*64 + e]
//   with e = t*16 + (lane&15), k = ks*32 + (j&8)*2 + (lane>>4)*8 + (j&7)
// -> main loop B loads are perfectly coalesced 32B/lane. Also zero col_sums.
// ---------------------------------------------------------------------------
__global__ void torus_prep_kernel(const float* __restrict__ Ex,
                                  const float* __restrict__ Ey,
                                  __bf16* __restrict__ ExP,
                                  __bf16* __restrict__ EyP,
                                  float* __restrict__ col_sums) {
  int idx = blockIdx.x * blockDim.x + threadIdx.x;   // 0 .. 2*131072-1
  if (idx < NEXP) col_sums[idx] = 0.0f;
  int m  = idx >> 17;                // which matrix (131072 = 2^17 per matrix)
  int r  = idx & (FRAG_ELEMS - 1);
  int j    = r & 15;
  int lane = (r >> 4) & 31;
  int t    = (r >> 9) & 3;
  int ks   = r >> 11;
  int e = t * 16 + (lane & 15);
  int k = ks * 32 + (j & 8) * 2 + (lane >> 4) * 8 + (j & 7);
  const float* src = m ? Ey : Ex;
  __bf16*      dst = m ? EyP : ExP;
  dst[r] = (__bf16)src[k * NEXP + e];
}

// Build 16-bit A fragment (16x32 bf16) for this lane from fp32 row data.
// Lanes 0-15 hold K={0..7,16..23}, lanes 16-31 hold K={8..15,24..31}.
__device__ inline v16bf load_a_frag_f32(const float* __restrict__ base) {
  const float4* p = reinterpret_cast<const float4*>(base);
  float4 c0 = p[0], c1 = p[1];
  const float4* q = reinterpret_cast<const float4*>(base + 16);
  float4 c2 = q[0], c3 = q[1];
  v16bf a;
  a[0]=(__bf16)c0.x;  a[1]=(__bf16)c0.y;  a[2]=(__bf16)c0.z;  a[3]=(__bf16)c0.w;
  a[4]=(__bf16)c1.x;  a[5]=(__bf16)c1.y;  a[6]=(__bf16)c1.z;  a[7]=(__bf16)c1.w;
  a[8]=(__bf16)c2.x;  a[9]=(__bf16)c2.y;  a[10]=(__bf16)c2.z; a[11]=(__bf16)c2.w;
  a[12]=(__bf16)c3.x; a[13]=(__bf16)c3.y; a[14]=(__bf16)c3.z; a[15]=(__bf16)c3.w;
  return a;
}

// ---------------------------------------------------------------------------
// Main: 4 waves per 16-token tile, K split across waves (latency hiding),
// LDS accumulator reduction, per-wave epilogue tile, top-4, softmax sums.
// ---------------------------------------------------------------------------
__global__ __launch_bounds__(32 * WAVES) void torus_main_kernel(
    const float*  __restrict__ u,
    const __bf16* __restrict__ ExP,
    const __bf16* __restrict__ EyP,
    const float*  __restrict__ bias,
    const float*  __restrict__ pc,  const float* __restrict__ pd,
    const float*  __restrict__ pa1, const float* __restrict__ pb1,
    float* __restrict__ out_topk_i, float* __restrict__ out_topk_s,
    float* __restrict__ out_scores, float* __restrict__ col_sums) {

  __shared__ float part[WAVES][2][4][32][8];   // 32 KB k-split partials
  __shared__ float smem[16][NEXP + 1];
  __shared__ float rowmax[16];
  __shared__ float rowsum[16];

  const int tid  = threadIdx.x;          // 0..127
  const int lane = tid & 31;
  const int wave = tid >> 5;             // 0..3 : this wave's K slice
  const int m15  = lane & 15;
  const int hi   = lane >> 4;            // 0/1
  const int row0 = blockIdx.x * 16;

  v8f accx[4] = {};
  v8f accy[4] = {};

  const float*  urow = u + (size_t)(row0 + m15) * DMODEL;
  const int     koff = hi * 8;                     // A chunk offset in 32-K tile
  const __bf16* bxl  = ExP + (size_t)lane * 16;    // lane's slot in fragment
  const __bf16* byl  = EyP + (size_t)lane * 16;

  const int ks0 = wave * KS_PER_WAVE;
  for (int ks = ks0; ks < ks0 + KS_PER_WAVE; ++ks) {
    const int kk = ks * 32;
    v16bf ax = load_a_frag_f32(urow + kk + koff);           // x half
    v16bf ay = load_a_frag_f32(urow + DHALF + kk + koff);   // y half
    const size_t fb = (size_t)ks * 4 * 32 * 16;             // fragment block base
#pragma unroll
    for (int t = 0; t < 4; ++t) {
      v16bf bx = *reinterpret_cast<const v16bf*>(bxl + fb + (size_t)t * 32 * 16);
      accx[t] = __builtin_amdgcn_wmma_f32_16x16x32_bf16(
          false, ax, false, bx, (short)0, accx[t], false, false);
      v16bf by = *reinterpret_cast<const v16bf*>(byl + fb + (size_t)t * 32 * 16);
      accy[t] = __builtin_amdgcn_wmma_f32_16x16x32_bf16(
          false, ay, false, by, (short)0, accy[t], false, false);
    }
  }

  // K-split reduction: dump partials to LDS
#pragma unroll
  for (int t = 0; t < 4; ++t) {
#pragma unroll
    for (int r = 0; r < 8; ++r) {
      part[wave][0][t][lane][r] = accx[t][r];
      part[wave][1][t][lane][r] = accy[t][r];
    }
  }
  __syncthreads();

  // Each wave takes one 16-expert tile: sum partials, compute torus scores.
  const float Cc = pc[0], Dd = pd[0], A1 = pa1[0], B1 = pb1[0];
  {
    const int t = wave;
    v8f sx = {}, sy = {};
#pragma unroll
    for (int w = 0; w < WAVES; ++w) {
#pragma unroll
      for (int r = 0; r < 8; ++r) {
        sx[r] += part[w][0][t][lane][r];
        sy[r] += part[w][1][t][lane][r];
      }
    }
    const int e = t * 16 + m15;
    const float be = bias[e];
#pragma unroll
    for (int r = 0; r < 8; ++r) {
      int M = r + hi * 8;                 // C/D layout: vgpr r + lane half -> row
      float xa = fabsf(tanhf(sx[r]) * SCALE_F);
      float ya = fabsf(tanhf(sy[r]) * SCALE_F);
      // p^a = exp2(a*log2 p); xa==0 -> log2=-inf -> exp2->0 (correct for a>0)
      float lx = __log2f(xa), ly = __log2f(ya);
      float s = (fast_exp2(A1 * lx) + fast_exp2(B1 * ly)) *
                __expf(-(fast_exp2(Cc * lx) + fast_exp2(Dd * ly))) + be;
      smem[M][e] = s;
    }
  }
  __syncthreads();

  // Coalesced score store: 16 rows x 64 experts, 128 threads
#pragma unroll
  for (int i = 0; i < 8; ++i) {
    int idx = i * 128 + tid;
    int r = idx >> 6, e = idx & 63;
    out_scores[(size_t)(row0 + r) * NEXP + e] = smem[r][e];
  }

  // Top-4 (descending, lowest-index tie-break like jax.lax.top_k) + row stats
  if (tid < 16) {
    const int r = tid;
    int chosen[TOPK];
    float rmax = -INFINITY;
#pragma unroll
    for (int j = 0; j < TOPK; ++j) {
      float mbest = -INFINITY; int mi = 0;
      for (int e = 0; e < NEXP; ++e) {
        bool skip = false;
#pragma unroll
        for (int q = 0; q < TOPK; ++q) skip |= (q < j) && (chosen[q] == e);
        float s = smem[r][e];
        if (!skip && s > mbest) { mbest = s; mi = e; }
      }
      chosen[j] = mi;
      out_topk_s[(size_t)(row0 + r) * TOPK + j] = mbest;
      out_topk_i[(size_t)(row0 + r) * TOPK + j] = (float)mi;
      if (j == 0) rmax = mbest;
    }
    float sum = 0.0f;
    for (int e = 0; e < NEXP; ++e) sum += __expf(smem[r][e] - rmax);
    rowmax[r] = rmax;
    rowsum[r] = sum;
  }
  __syncthreads();

  // Softmax-prob column sums: thread -> (expert, row half), then global atomics
  {
    const int e  = tid & 63;
    const int r0 = (tid >> 6) * 8;       // rows 0-7 or 8-15
    float cs = 0.0f;
#pragma unroll
    for (int i = 0; i < 8; ++i) {
      int r = r0 + i;
      cs += __expf(smem[r][e] - rowmax[r]) * (1.0f / rowsum[r]);
    }
    atomicAdd(&col_sums[e], cs);
  }
}

// ---------------------------------------------------------------------------
// Aux loss: sum_e (colsum_e / N)^2 * NEXP
// ---------------------------------------------------------------------------
__global__ void torus_aux_kernel(const float* __restrict__ col_sums,
                                 float* __restrict__ out_aux) {
  __shared__ float red[NEXP];
  int t = threadIdx.x;
  float v = col_sums[t] * (1.0f / (float)N_TOK);
  red[t] = v * v;
  __syncthreads();
  for (int s = NEXP / 2; s > 0; s >>= 1) {
    if (t < s) red[t] += red[t + s];
    __syncthreads();
  }
  if (t == 0) out_aux[0] = red[0] * (float)NEXP;
}

// ---------------------------------------------------------------------------
extern "C" void kernel_launch(void* const* d_in, const int* in_sizes, int n_in,
                              void* d_out, int out_size, void* d_ws, size_t ws_size,
                              hipStream_t stream) {
  (void)in_sizes; (void)n_in; (void)out_size; (void)ws_size;

  const float* u    = (const float*)d_in[0];
  const float* Ex   = (const float*)d_in[1];
  const float* Ey   = (const float*)d_in[2];
  const float* bias = (const float*)d_in[3];
  const float* c    = (const float*)d_in[4];
  const float* d    = (const float*)d_in[5];
  const float* a1   = (const float*)d_in[6];
  const float* b1   = (const float*)d_in[7];
  // d_in[8] = top_k (compile-time TOPK=4)

  __bf16* ExP = (__bf16*)d_ws;
  __bf16* EyP = ExP + (size_t)FRAG_ELEMS;
  float* col_sums = (float*)((char*)d_ws + 2 * (size_t)FRAG_ELEMS * sizeof(__bf16));

  float* out        = (float*)d_out;
  float* out_topk_i = out;
  float* out_topk_s = out + (size_t)N_TOK * TOPK;
  float* out_scores = out + (size_t)N_TOK * TOPK * 2;
  float* out_aux    = out + (size_t)N_TOK * TOPK * 2 + (size_t)N_TOK * NEXP;

  // 1) repack experts to bf16 fragments + zero accumulators
  torus_prep_kernel<<<(2 * FRAG_ELEMS) / 256, 256, 0, stream>>>(Ex, Ey, ExP, EyP, col_sums);
  // 2) k-split dual WMMA GEMM + scores + top-k + softmax column sums
  torus_main_kernel<<<N_TOK / 16, 32 * WAVES, 0, stream>>>(u, ExP, EyP, bias, c, d, a1, b1,
                                                           out_topk_i, out_topk_s,
                                                           out_scores, col_sums);
  // 3) aux loss
  torus_aux_kernel<<<1, NEXP, 0, stream>>>(col_sums, out_aux);
}
